// SelfAttentionLayer_85985245266330
// MI455X (gfx1250) — matrix operand order repaired and checked by reading, not compiled, not run on hardware
//
#include <hip/hip_runtime.h>
#include <hip/hip_bf16.h>

typedef __attribute__((ext_vector_type(16))) __bf16 bf16x16;
typedef __attribute__((ext_vector_type(8)))  float  f32x8;

#define D_MODEL 1024
#define NUM_HEADS 16
#define DEPTH 64
#define SEQ 2048
#define BATCH 4
#define M_ROWS (BATCH * SEQ)          // 8192
#define BH (BATCH * NUM_HEADS)        // 64

// ---------------------------------------------------------------------------
// CDNA5 async global->LDS copy (ASYNCcnt path).  lds_off = byte offset into
// LDS (low 32 bits of the generic/shared address, per ISA 10.2 aperture rule).
// ---------------------------------------------------------------------------
__device__ __forceinline__ void async_copy_b128(unsigned lds_off, const void* gaddr) {
    asm volatile("global_load_async_to_lds_b128 %0, %1, off"
                 :: "v"(lds_off), "v"(gaddr) : "memory");
}
__device__ __forceinline__ void wait_async0() {
    asm volatile("s_wait_asynccnt 0" ::: "memory");
}

// ---------------------------------------------------------------------------
// f32 -> bf16 elementwise convert
// ---------------------------------------------------------------------------
__global__ void sa_conv_bf16(const float* __restrict__ src, __bf16* __restrict__ dst, int n) {
    int i = blockIdx.x * blockDim.x + threadIdx.x;
    int stride = gridDim.x * blockDim.x;
    for (; i < n; i += stride) dst[i] = (__bf16)src[i];
}

// ---------------------------------------------------------------------------
// W [K x N] f32  ->  Wt [N x K] bf16  (LDS tile transpose, 32x32)
// ---------------------------------------------------------------------------
__global__ void sa_convWT(const float* __restrict__ W, __bf16* __restrict__ Wt) {
    __shared__ float tile[32][33];
    int n0 = blockIdx.x * 32;
    int k0 = blockIdx.y * 32;
    int tx = threadIdx.x & 31;
    int ty = threadIdx.x >> 5;           // 0..7
#pragma unroll
    for (int i = 0; i < 4; ++i) {
        int r = ty + i * 8;
        tile[r][tx] = W[(size_t)(k0 + r) * D_MODEL + n0 + tx];
    }
    __syncthreads();
#pragma unroll
    for (int i = 0; i < 4; ++i) {
        int r = ty + i * 8;
        Wt[(size_t)(n0 + r) * D_MODEL + k0 + tx] = (__bf16)tile[tx][r];
    }
}

// ---------------------------------------------------------------------------
// GEMM: C[M x 1024] = A[M x 1024](bf16) * W(bf16, stored transposed [n][k]) + bias
// mode 0: write bf16 head-split  [b][h][s][64]      (Q, K projections)
// mode 1: write f32 row-major    [m][n] to d_out    (output projection)
// mode 2: write bf16 head-split TRANSPOSED [b][h][64][s]  (V projection)
// Block: 256 threads = 8 waves. Tile: BM=128 (wave-row 16), BN=64, BK=32.
// A/B tiles staged with global_load_async_to_lds_b128 (ASYNCcnt).
// ---------------------------------------------------------------------------
__global__ void __launch_bounds__(256)
sa_gemm(const __bf16* __restrict__ A, const __bf16* __restrict__ Wt,
        const float* __restrict__ bias, void* __restrict__ dst, int mode) {
    __shared__ unsigned int As[128 * 16];   // 128 rows x 32 bf16 (as dwords), 8 KB
    __shared__ unsigned int Bs[64 * 16];    // 64 n-rows x 32 bf16 (as dwords), 4 KB
    const __bf16* Asb = (const __bf16*)As;
    const __bf16* Bsb = (const __bf16*)Bs;
    const unsigned As_off = (unsigned)(size_t)(void*)As;   // LDS byte offset
    const unsigned Bs_off = (unsigned)(size_t)(void*)Bs;

    const int m0 = blockIdx.x * 128;
    const int n0 = blockIdx.y * 64;
    const int tid = threadIdx.x;
    const int lane = tid & 31;
    const int wave = tid >> 5;
    const int half = lane >> 4;
    const int nn = lane & 15;

    const unsigned int* Au = (const unsigned int*)A;   // full row = 512 dwords
    const unsigned int* Bu = (const unsigned int*)Wt;

    f32x8 ct[4];
#pragma unroll
    for (int t = 0; t < 4; ++t) ct[t] = (f32x8){};

    for (int kt = 0; kt < D_MODEL / 32; ++kt) {
        // ---- async-stage A tile: 512 x 16B chunks (4 chunks per 32-elem row) ----
#pragma unroll
        for (int it = 0; it < 2; ++it) {
            int c = tid + it * 256;
            int row = c >> 2, c16 = c & 3;
            async_copy_b128(As_off + c * 16,
                            &Au[(size_t)(m0 + row) * 512 + kt * 16 + c16 * 4]);
        }
        // ---- async-stage B tile: 256 x 16B chunks ----
        {
            int row = tid >> 2, c16 = tid & 3;
            async_copy_b128(Bs_off + tid * 16,
                            &Bu[(size_t)(n0 + row) * 512 + kt * 16 + c16 * 4]);
        }
        if (kt + 1 < D_MODEL / 32) {
            __builtin_prefetch(&Au[(size_t)(m0 + (tid >> 1)) * 512 + (kt + 1) * 16], 0, 3);
        }
        wait_async0();
        __syncthreads();

        // A fragment: rows wave*16 .. +15  (contiguous k = 16*half + s)
        bf16x16 a;
        {
            int rowbase = (wave * 16 + nn) * 32;
#pragma unroll
            for (int j = 0; j < 8; ++j) {
                int kl = 2 * j + half * 16;
                a[2 * j]     = Asb[rowbase + kl];
                a[2 * j + 1] = Asb[rowbase + kl + 1];
            }
        }
#pragma unroll
        for (int t = 0; t < 4; ++t) {
            bf16x16 b;
            int nb = (t * 16 + nn) * 32 + half * 16;
#pragma unroll
            for (int j = 0; j < 8; ++j) {
                b[2 * j]     = Bsb[nb + 2 * j];
                b[2 * j + 1] = Bsb[nb + 2 * j + 1];
            }
            ct[t] = __builtin_amdgcn_wmma_f32_16x16x32_bf16(
                false, a, false, b, (short)0, ct[t], false, false);
        }
        __syncthreads();
    }

    // epilogue
#pragma unroll
    for (int t = 0; t < 4; ++t) {
#pragma unroll
        for (int r = 0; r < 8; ++r) {
            int M = m0 + wave * 16 + r + 8 * half;
            int N = n0 + t * 16 + nn;
            float v = ct[t][r] + bias[N];
            int b = M >> 11, s = M & 2047;
            int h = N >> 6,  d = N & 63;
            if (mode == 0) {
                ((__bf16*)dst)[(((size_t)b * NUM_HEADS + h) * SEQ + s) * DEPTH + d] = (__bf16)v;
            } else if (mode == 2) {   // V: per-head transposed [bh][d][s]
                ((__bf16*)dst)[(((size_t)b * NUM_HEADS + h) * DEPTH + d) * SEQ + s] = (__bf16)v;
            } else {
                ((float*)dst)[(size_t)M * D_MODEL + N] = v;
            }
        }
    }
}

// ---------------------------------------------------------------------------
// Attention: one block = one (b*h, 16-row q tile).
//   1) logits(16 x 2048) = Q Ktile^T / 8  via WMMA -> LDS (128 KB)
//   2) softmax per row (wave shuffle reduce); write attn f32 to HBM once
//      (contiguous 16B chunks per lane), overwrite LDS logits with P
//   3) out(16 x 64) = P * Vt via WMMA (Vt fragments are contiguous 32B/lane),
//      ds_add_f32 cross-wave reduce, write bf16 into concat buffer
// ---------------------------------------------------------------------------
__global__ void __launch_bounds__(256)
sa_attention(const __bf16* __restrict__ Qbf, const __bf16* __restrict__ Kbf,
             const __bf16* __restrict__ Vt, float* __restrict__ attn,
             __bf16* __restrict__ Cbf) {
    __shared__ float logits[16][SEQ];    // 128 KB
    __shared__ float outacc[16][DEPTH];  // 4 KB

    const int qt = blockIdx.x;           // 0..127
    const int bh = blockIdx.y;           // 0..63
    const int q0 = qt * 16;
    const int tid = threadIdx.x;
    const int lane = tid & 31;
    const int wave = tid >> 5;
    const int half = lane >> 4;
    const int nn = lane & 15;

    const __bf16* Qrow  = Qbf + ((size_t)bh * SEQ + q0) * DEPTH;
    const __bf16* Kbase = Kbf + (size_t)bh * SEQ * DEPTH;
    const __bf16* Vbase = Vt  + (size_t)bh * DEPTH * SEQ;   // [d][key]

    // ---- Step 1: logits = Q K^T * (1/sqrt(64)) ----
    bf16x16 aq0, aq1;   // Q fragments for d = 0..31 and 32..63
#pragma unroll
    for (int j = 0; j < 8; ++j) {
        int kl = 2 * j + half * 16;
        aq0[2 * j]     = Qrow[nn * DEPTH + kl];
        aq0[2 * j + 1] = Qrow[nn * DEPTH + kl + 1];
        aq1[2 * j]     = Qrow[nn * DEPTH + 32 + kl];
        aq1[2 * j + 1] = Qrow[nn * DEPTH + 32 + kl + 1];
    }
    for (int j = wave; j < SEQ / 16; j += 8) {
        int key0 = j * 16;
        f32x8 acc = (f32x8){};
        bf16x16 b0, b1;   // K^T fragments: B[d][key]; 32B contiguous per lane
        const __bf16* kp = Kbase + (size_t)(key0 + nn) * DEPTH + half * 16;
#pragma unroll
        for (int jj = 0; jj < 16; ++jj) {
            b0[jj] = kp[jj];
            b1[jj] = kp[32 + jj];
        }
        acc = __builtin_amdgcn_wmma_f32_16x16x32_bf16(false, aq0, false, b0, (short)0, acc, false, false);
        acc = __builtin_amdgcn_wmma_f32_16x16x32_bf16(false, aq1, false, b1, (short)0, acc, false, false);
#pragma unroll
        for (int r = 0; r < 8; ++r)
            logits[r + 8 * half][key0 + nn] = acc[r] * 0.125f;   // 1/sqrt(64)
    }
    __syncthreads();

    // ---- Step 2: softmax rows {wave, wave+8}; write attn; overwrite LDS with P ----
#pragma unroll
    for (int rr = 0; rr < 2; ++rr) {
        int r = wave + rr * 8;
        float m = -__builtin_inff();
        for (int c = lane * 4; c < SEQ; c += 128) {
#pragma unroll
            for (int i = 0; i < 4; ++i) m = fmaxf(m, logits[r][c + i]);
        }
#pragma unroll
        for (int off = 16; off > 0; off >>= 1) m = fmaxf(m, __shfl_xor(m, off, 32));
        float s = 0.f;
        for (int c = lane * 4; c < SEQ; c += 128) {
#pragma unroll
            for (int i = 0; i < 4; ++i) s += __expf(logits[r][c + i] - m);
        }
#pragma unroll
        for (int off = 16; off > 0; off >>= 1) s += __shfl_xor(s, off, 32);
        float inv = 1.0f / s;
        float* arow = attn + ((size_t)bh * SEQ + q0 + r) * SEQ;
        for (int c = lane * 4; c < SEQ; c += 128) {
#pragma unroll
            for (int i = 0; i < 4; ++i) {
                float p = __expf(logits[r][c + i] - m) * inv;
                arow[c + i] = p;          // merges to global_store_b128
                logits[r][c + i] = p;
            }
        }
    }
    for (int idx = tid; idx < 16 * DEPTH; idx += 256)
        outacc[idx >> 6][idx & 63] = 0.f;
    __syncthreads();

    // ---- Step 3: out = P * Vt.  wave -> (ntile = wave&3, key half = wave>>2) ----
    {
        int d0 = (wave & 3) * 16;
        int keybase = (wave >> 2) * (SEQ / 2);
        const __bf16* vrow = Vbase + (size_t)(d0 + nn) * SEQ;   // column d0+nn of V
        f32x8 acc = (f32x8){};
        for (int kk = 0; kk < (SEQ / 2) / 32; ++kk) {
            int key0 = keybase + kk * 32;
            bf16x16 a, b;
            const __bf16* vp = vrow + key0 + half * 16;          // 32B contiguous
#pragma unroll
            for (int j = 0; j < 8; ++j) {
                int kl = 2 * j + half * 16;
                a[2 * j]     = (__bf16)logits[nn][key0 + kl];
                a[2 * j + 1] = (__bf16)logits[nn][key0 + kl + 1];
                b[2 * j]     = vp[2 * j];
                b[2 * j + 1] = vp[2 * j + 1];
            }
            acc = __builtin_amdgcn_wmma_f32_16x16x32_bf16(false, a, false, b, (short)0, acc, false, false);
        }
#pragma unroll
        for (int r = 0; r < 8; ++r)
            atomicAdd(&outacc[r + 8 * half][d0 + nn], acc[r]);   // ds_add_f32
    }
    __syncthreads();

    // write concat tile: Cbf[(b*2048 + s)][h*64 + d]
    int b = bh >> 4, h = bh & 15;
    for (int idx = tid; idx < 16 * DEPTH; idx += 256) {
        int row = idx >> 6, d = idx & 63;
        Cbf[((size_t)b * SEQ + q0 + row) * D_MODEL + h * DEPTH + d] = (__bf16)outacc[row][d];
    }
}

// ---------------------------------------------------------------------------
extern "C" void kernel_launch(void* const* d_in, const int* in_sizes, int n_in,
                              void* d_out, int out_size, void* d_ws, size_t ws_size,
                              hipStream_t stream) {
    (void)in_sizes; (void)n_in; (void)out_size; (void)ws_size;
    const float* X  = (const float*)d_in[0];
    const float* wq = (const float*)d_in[1];
    const float* bq = (const float*)d_in[2];
    const float* wk = (const float*)d_in[3];
    const float* bk = (const float*)d_in[4];
    const float* wv = (const float*)d_in[5];
    const float* bv = (const float*)d_in[6];
    const float* wo = (const float*)d_in[7];
    const float* bo = (const float*)d_in[8];

    float* out  = (float*)d_out;                               // [4,2048,1024]
    float* attn = out + (size_t)M_ROWS * D_MODEL;              // [4,16,2048,2048]

    char* ws = (char*)d_ws;
    size_t off = 0;
    auto carve = [&](size_t bytes) { void* p = ws + off; off += (bytes + 255) & ~(size_t)255; return p; };
    const size_t actBytes = (size_t)M_ROWS * D_MODEL * sizeof(__bf16);   // 16 MB
    const size_t wBytes   = (size_t)D_MODEL * D_MODEL * sizeof(__bf16);  // 2 MB
    __bf16* Xbf = (__bf16*)carve(actBytes);
    __bf16* Wqt = (__bf16*)carve(wBytes);
    __bf16* Wkt = (__bf16*)carve(wBytes);
    __bf16* Wvt = (__bf16*)carve(wBytes);
    __bf16* Wot = (__bf16*)carve(wBytes);
    __bf16* Qbf = (__bf16*)carve(actBytes);
    __bf16* Kbf = (__bf16*)carve(actBytes);
    __bf16* Vtb = (__bf16*)carve(actBytes);                    // V transposed per head
    __bf16* Cbf = (__bf16*)carve(actBytes);

    // 1) convert activations + weights to bf16 (weights transposed)
    sa_conv_bf16<<<4096, 256, 0, stream>>>(X, Xbf, M_ROWS * D_MODEL);
    dim3 wtGrid(32, 32);
    sa_convWT<<<wtGrid, 256, 0, stream>>>(wq, Wqt);
    sa_convWT<<<wtGrid, 256, 0, stream>>>(wk, Wkt);
    sa_convWT<<<wtGrid, 256, 0, stream>>>(wv, Wvt);
    sa_convWT<<<wtGrid, 256, 0, stream>>>(wo, Wot);

    // 2) QKV projections (WMMA); V written per-head transposed for PV fragments
    dim3 gGrid(M_ROWS / 128, D_MODEL / 64);
    sa_gemm<<<gGrid, 256, 0, stream>>>(Xbf, Wqt, bq, (void*)Qbf, 0);
    sa_gemm<<<gGrid, 256, 0, stream>>>(Xbf, Wkt, bk, (void*)Kbf, 0);
    sa_gemm<<<gGrid, 256, 0, stream>>>(Xbf, Wvt, bv, (void*)Vtb, 2);

    // 3) attention (WMMA logits + softmax + WMMA PV), writes attn once
    dim3 aGrid(SEQ / 16, BH);
    sa_attention<<<aGrid, 256, 0, stream>>>(Qbf, Kbf, Vtb, attn, Cbf);

    // 4) output projection (WMMA), f32 result
    sa_gemm<<<gGrid, 256, 0, stream>>>(Cbf, Wot, bo, (void*)out, 1);
}
